// MixtureOfExperts_38319698215038
// MI455X (gfx1250) — matrix operand order repaired and checked by reading, not compiled, not run on hardware
//
#include <hip/hip_runtime.h>
#include <hip/hip_bf16.h>
#include <hip/hip_fp16.h>

#define H    1024
#define F    4096
#define E    16
#define TOPK 4
#define T    2048
#define MR   64         // token rows per expert-FFN block (4 M-tiles)
#define MT64 (T / MR)   // max 64-token tiles per expert
#define FC   128        // F-chunk per fused iteration (expert kernel)
#define KS1  128        // phase-1 K slab

// LDS row strides (f16 elements); all chosen so rows are 16B-aligned
#define AX_S  (H + 8)    // 1032 -> 2064 B/row
#define BS1_S 136        // 272 B/row
#define BS2_S 40         // 80 B/row
#define HH_S  (FC + 8)   // 136 -> 272 B/row

typedef __attribute__((ext_vector_type(16))) _Float16 v16h;
typedef __attribute__((ext_vector_type(8)))  float    v8f;

// ---------------------------------------------------------------------------
// WMMA fragment helpers (16-bit operands; contiguous k order chosen here, same for A and B).
// A 16x32 f16: lane l -> M=l%16; element s: K=16*(l/16)+s (contiguous run, same order as B)
// B 32x16 f16 staged transposed in LDS: lane l -> N=l%16, K=16*(l/16)+q (contiguous run)
// ---------------------------------------------------------------------------
__device__ __forceinline__ v16h load_a_frag(const _Float16* row, int kbase, int lanehi) {
  v16h a;
#pragma unroll
  for (int v = 0; v < 8; ++v) {
    const int kk = kbase + 2 * v + 16 * lanehi;
    a[2 * v]     = row[kk];
    a[2 * v + 1] = row[kk + 1];
  }
  return a;
}

__device__ __forceinline__ v16h load_b_frag(const _Float16* p) {
  v16h b;
#pragma unroll
  for (int q = 0; q < 16; ++q) b[q] = p[q];
  return b;
}

// Async gather of one 16-byte chunk from global f16 into LDS (CDNA5 async path).
__device__ __forceinline__ void async_g2l_b128(_Float16* ldsDst, const _Float16* gSrc) {
  const unsigned ldsOff = (unsigned)(size_t)ldsDst;  // low 32 bits of the pointer are the LDS address
  asm volatile("global_load_async_to_lds_b128 %0, %1, off"
               :: "v"(ldsOff), "v"(gSrc) : "memory");
}
__device__ __forceinline__ void wait_async0() {
  asm volatile("s_wait_asynccnt 0x0" ::: "memory");
}

// ---------------------------------------------------------------------------
// Kernel 1: f16 copy of x (feeds all WMMA A-matrices).
// ---------------------------------------------------------------------------
__global__ __launch_bounds__(256) void moe_cvt_f16(const float* __restrict__ x,
                                                   _Float16* __restrict__ xh) {
  const int i = blockIdx.x * 256 + threadIdx.x;
  xh[i] = (_Float16)x[i];
}

// ---------------------------------------------------------------------------
// Kernel 2: hh16 = relu(x @ Wr1 + br1), WMMA f16.
// Block = 8 waves; tile = 16 tokens x 256 cols; wave owns 32 cols (2 C-tiles).
// ---------------------------------------------------------------------------
__global__ __launch_bounds__(256, 1) void moe_router_gemm1(
    const _Float16* __restrict__ xh, const float* __restrict__ Wr1,
    const float* __restrict__ br1, _Float16* __restrict__ hh16) {
  extern __shared__ _Float16 smem[];
  _Float16* Axr = smem;                 // 16 x AX_S
  _Float16* Bsr = Axr + 16 * AX_S;      // 256 x BS1_S  ([n][k] transposed slab)

  const int mt = blockIdx.x;            // token tile
  const int nb = blockIdx.y;            // 256-col block
  const int tid = threadIdx.x;
  const int wave = tid >> 5, lane = tid & 31;
  const int lanelo = lane & 15, lanehi = lane >> 4;

  for (int idx = tid; idx < 16 * (H / 8); idx += 256) {
    const int r = idx >> 7, c8 = idx & 127;
    async_g2l_b128(&Axr[r * AX_S + c8 * 8],
                   xh + ((size_t)(mt * 16 + r)) * H + c8 * 8);
  }
  wait_async0();
  __syncthreads();

  v8f acc0 = {}, acc1 = {};
  for (int ktb = 0; ktb < H; ktb += 128) {
    __syncthreads();
    for (int idx = tid; idx < 128 * 256; idx += 256) {   // coalesced over n
      const int k = idx >> 8, n = idx & 255;
      Bsr[n * BS1_S + k] = (_Float16)Wr1[(size_t)(ktb + k) * H + nb * 256 + n];
    }
    __syncthreads();
    const _Float16* arow = Axr + (size_t)lanelo * AX_S;
#pragma unroll
    for (int ki = 0; ki < 128; ki += 32) {
      const v16h a = load_a_frag(arow, ktb + ki, lanehi);
      const v16h b0 = load_b_frag(&Bsr[(wave * 32 + lanelo) * BS1_S + ki + 16 * lanehi]);
      acc0 = __builtin_amdgcn_wmma_f32_16x16x32_f16(false, a, false, b0,
                                                    (short)0, acc0, false, false);
      const v16h b1 = load_b_frag(&Bsr[(wave * 32 + 16 + lanelo) * BS1_S + ki + 16 * lanehi]);
      acc1 = __builtin_amdgcn_wmma_f32_16x16x32_f16(false, a, false, b1,
                                                    (short)0, acc1, false, false);
    }
  }
  const int col0 = nb * 256 + wave * 32 + lanelo;
  const float bz0 = br1[col0], bz1 = br1[col0 + 16];
#pragma unroll
  for (int v = 0; v < 8; ++v) {
    const int t = mt * 16 + v + 8 * lanehi;
    hh16[(size_t)t * H + col0]      = (_Float16)fmaxf(acc0[v] + bz0, 0.0f);
    hh16[(size_t)t * H + col0 + 16] = (_Float16)fmaxf(acc1[v] + bz1, 0.0f);
  }
}

// ---------------------------------------------------------------------------
// Kernel 3: logits = hh16 @ Wr2 + br2 ; top-4 + softmax per token.
// ---------------------------------------------------------------------------
__global__ __launch_bounds__(256) void moe_router_topk(
    const _Float16* __restrict__ hh16, const float* __restrict__ Wr2,
    const float* __restrict__ br2, int* __restrict__ topi,
    float* __restrict__ topw) {
  const int t = blockIdx.x * blockDim.x + threadIdx.x;
  if (t >= T) return;
  float acc[E];
#pragma unroll
  for (int e = 0; e < E; ++e) acc[e] = br2[e];
  const _Float16* hr = hh16 + (size_t)t * H;
  for (int k = 0; k < H; ++k) {
    const float hv = (float)hr[k];
    const float* w = Wr2 + (size_t)k * E;
#pragma unroll
    for (int e = 0; e < E; ++e) acc[e] = fmaf(hv, w[e], acc[e]);
  }
  int idx[TOPK]; float val[TOPK];
  bool used[E];
#pragma unroll
  for (int e = 0; e < E; ++e) used[e] = false;
#pragma unroll
  for (int s = 0; s < TOPK; ++s) {
    int bi = 0; float bv = -3.4e38f;
#pragma unroll
    for (int e = 0; e < E; ++e)
      if (!used[e] && acc[e] > bv) { bv = acc[e]; bi = e; }
    used[bi] = true; idx[s] = bi; val[s] = bv;
  }
  const float m = val[0];
  float se = 0.0f;
#pragma unroll
  for (int s = 0; s < TOPK; ++s) { val[s] = __expf(val[s] - m); se += val[s]; }
  const float inv = 1.0f / se;
#pragma unroll
  for (int s = 0; s < TOPK; ++s) {
    topi[t * TOPK + s] = idx[s];
    topw[t * TOPK + s] = val[s] * inv;
  }
}

// ---------------------------------------------------------------------------
// Kernel 4: deterministic per-expert ordered token lists (no atomics).
// Lists padded to MR so the expert kernel can use 64-row tiles.
// ---------------------------------------------------------------------------
__global__ void moe_build_lists(const int* __restrict__ topi,
                                int* __restrict__ counts,
                                int* __restrict__ lists,
                                int* __restrict__ slots) {
  const int e = blockIdx.x;
  if (threadIdx.x != 0) return;
  int cnt = 0;
  for (int t = 0; t < T; ++t) {
#pragma unroll
    for (int k = 0; k < TOPK; ++k) {
      if (topi[t * TOPK + k] == e) {
        lists[e * T + cnt] = t;
        slots[e * T + cnt] = k;
        ++cnt;
      }
    }
  }
  counts[e] = cnt;
  int pad = (cnt + MR - 1) & ~(MR - 1);
  if (pad > T) pad = T;
  for (int i = cnt; i < pad; ++i) { lists[e * T + i] = -1; slots[e * T + i] = 0; }
}

// ---------------------------------------------------------------------------
// Kernel 5: fused expert FFN (WMMA f16), 64 gathered tokens per block.
// Block = 32 waves (1024 thr). Wave w -> msub = w>>3 (16-row subtile),
// wsl = w&7 (16-col h-slice in phase 1, 128-col output slice in phase 2).
// ---------------------------------------------------------------------------
__global__ __launch_bounds__(1024, 1) void moe_expert_ffn(
    const _Float16* __restrict__ xh, const float* __restrict__ W1,
    const float* __restrict__ b1, const float* __restrict__ W2,
    const float* __restrict__ b2, const int* __restrict__ counts,
    const int* __restrict__ lists, const int* __restrict__ slots,
    float* __restrict__ ybuf) {
  const int e  = blockIdx.x / MT64;
  const int mt = blockIdx.x % MT64;
  const int ne = counts[e];
  if (mt * MR >= ne) return;   // uniform across block

  extern __shared__ _Float16 smem[];
  _Float16* Ax  = smem;                    // MR  x AX_S   (gathered activations)
  _Float16* Bs1 = Ax  + MR * AX_S;         // 128 x BS1_S  (W1 k-slab, [n][k])
  _Float16* Bs2 = Bs1 + 128 * BS1_S;       // H   x BS2_S  (W2 k-slab, [n][k])
  _Float16* Hh  = Bs2 + (size_t)H * BS2_S; // MR  x HH_S   (gelu(h) chunk)
  int* tlds = (int*)(Hh + MR * HH_S);
  int* slds = tlds + MR;

  const int tid = threadIdx.x;
  const int wave = tid >> 5, lane = tid & 31;
  const int lanelo = lane & 15, lanehi = lane >> 4;
  const int msub = wave >> 3;   // 0..3: 16-row subtile
  const int wsl  = wave & 7;    // 0..7: col slice

  if (tid < MR) {
    const int gi = mt * MR + tid;
    tlds[tid] = (gi < ne) ? lists[e * T + gi] : -1;
    slds[tid] = (gi < ne) ? slots[e * T + gi] : 0;
  }
  __syncthreads();

  // async gather of token rows into LDS (f16, 16B granules)
  for (int idx = tid; idx < MR * (H / 8); idx += 1024) {
    const int r = idx >> 7, c8 = idx & 127;
    const int tok = tlds[r];
    _Float16* dst = &Ax[r * AX_S + c8 * 8];
    if (tok >= 0) {
      async_g2l_b128(dst, xh + (size_t)tok * H + c8 * 8);
    } else {
#pragma unroll
      for (int q = 0; q < 8; ++q) dst[q] = (_Float16)0.0f;
    }
  }
  wait_async0();
  __syncthreads();

  v8f acc[8];
  {
    const v8f zero = {};
#pragma unroll
    for (int j = 0; j < 8; ++j) acc[j] = zero;
  }

  const float* W1e = W1 + (size_t)e * H * F;
  const float* W2e = W2 + (size_t)e * F * H;

  for (int fcBase = 0; fcBase < F; fcBase += FC) {
    // prefetch next chunk's weights toward L2 (global_prefetch_b8)
    if (fcBase + FC < F) {
      if (tid < 32)
        __builtin_prefetch(W1e + (size_t)(tid * 32) * F + (fcBase + FC), 0, 1);
      else if (tid < 64)
        __builtin_prefetch(W2e + (size_t)(fcBase + FC + (tid - 32) * 4) * H, 0, 1);
    }

    // ---- Phase 1: h-tile [16 x 16] per wave, K = H in slabs of KS1 ----
    v8f hacc = {};
    for (int ktb = 0; ktb < H; ktb += KS1) {
      __syncthreads();                                   // Bs1 reuse guard
      for (int idx = tid; idx < KS1 * 128; idx += 1024) {  // coalesced over n
        const int k = idx >> 7, n = idx & 127;
        Bs1[n * BS1_S + k] = (_Float16)W1e[(size_t)(ktb + k) * F + fcBase + n];
      }
      __syncthreads();
      const _Float16* arow = Ax + (size_t)(msub * 16 + lanelo) * AX_S;
#pragma unroll
      for (int ki = 0; ki < KS1; ki += 32) {
        const v16h a = load_a_frag(arow, ktb + ki, lanehi);
        const v16h b = load_b_frag(&Bs1[(wsl * 16 + lanelo) * BS1_S + ki + 16 * lanehi]);
        hacc = __builtin_amdgcn_wmma_f32_16x16x32_f16(false, a, false, b,
                                                      (short)0, hacc, false, false);
      }
    }
    const int ncol1 = fcBase + wsl * 16;
    const float bias1 = b1[(size_t)e * F + ncol1 + lanelo];
#pragma unroll
    for (int v = 0; v < 8; ++v) {
      float hv = hacc[v] + bias1;
      hv = 0.5f * hv * (1.0f + erff(hv * 0.70710678118f));  // exact GELU
      Hh[(msub * 16 + v + 8 * lanehi) * HH_S + wsl * 16 + lanelo] = (_Float16)hv;
    }
    __syncthreads();

    // ---- Phase 2: acc += Hh[MR x FC] @ W2e[chunk, :], K slabs of 32 ----
    for (int ks = 0; ks < FC; ks += 32) {
      for (int idx = tid; idx < 32 * H; idx += 1024) {     // coalesced over n
        const int k = idx >> 10, n = idx & 1023;
        Bs2[n * BS2_S + k] = (_Float16)W2e[(size_t)(fcBase + ks + k) * H + n];
      }
      __syncthreads();
      const v16h a = load_a_frag(Hh + (size_t)(msub * 16 + lanelo) * HH_S, ks, lanehi);
#pragma unroll
      for (int j = 0; j < 8; ++j) {
        const int ncol = wsl * 128 + j * 16;
        const v16h b = load_b_frag(&Bs2[(ncol + lanelo) * BS2_S + 16 * lanehi]);
        acc[j] = __builtin_amdgcn_wmma_f32_16x16x32_f16(false, a, false, b,
                                                        (short)0, acc[j], false, false);
      }
      __syncthreads();
    }
  }

  // ---- epilogue: +b2, scatter to per-(token,slot) buffer ----
#pragma unroll
  for (int j = 0; j < 8; ++j) {
    const int ncol = wsl * 128 + j * 16 + lanelo;
    const float bias2 = b2[(size_t)e * H + ncol];
#pragma unroll
    for (int v = 0; v < 8; ++v) {
      const int row = msub * 16 + v + 8 * lanehi;  // C-layout: VGPR v -> M = v (+8)
      const int tok = tlds[row];
      if (tok >= 0) {
        ybuf[((size_t)tok * TOPK + slds[row]) * H + ncol] = acc[j][v] + bias2;
      }
    }
  }
}

// ---------------------------------------------------------------------------
// Kernel 6: out[t,h] = sum_k topw[t,k] * ybuf[t,k,h]
// ---------------------------------------------------------------------------
__global__ __launch_bounds__(256) void moe_combine(
    const float* __restrict__ ybuf, const float* __restrict__ topw,
    float* __restrict__ out) {
  const int i = blockIdx.x * blockDim.x + threadIdx.x;
  if (i >= T * H) return;
  const int t = i >> 10, h = i & (H - 1);
  float s = 0.0f;
#pragma unroll
  for (int k = 0; k < TOPK; ++k)
    s = fmaf(topw[t * TOPK + k], ybuf[((size_t)t * TOPK + k) * H + h], s);
  out[i] = s;
}

// ---------------------------------------------------------------------------
extern "C" void kernel_launch(void* const* d_in, const int* in_sizes, int n_in,
                              void* d_out, int out_size, void* d_ws, size_t ws_size,
                              hipStream_t stream) {
  (void)in_sizes; (void)n_in; (void)out_size; (void)ws_size;
  const float* x   = (const float*)d_in[0];
  const float* Wr1 = (const float*)d_in[1];
  const float* br1 = (const float*)d_in[2];
  const float* Wr2 = (const float*)d_in[3];
  const float* br2 = (const float*)d_in[4];
  const float* W1  = (const float*)d_in[5];
  const float* b1  = (const float*)d_in[6];
  const float* W2  = (const float*)d_in[7];
  const float* b2  = (const float*)d_in[8];

  char* ws = (char*)d_ws;
  _Float16* xh     = (_Float16*)ws;  ws += (size_t)T * H * 2;
  _Float16* hh16   = (_Float16*)ws;  ws += (size_t)T * H * 2;
  int*      topi   = (int*)ws;       ws += (size_t)T * TOPK * 4;
  float*    topw   = (float*)ws;     ws += (size_t)T * TOPK * 4;
  int*      counts = (int*)ws;       ws += 256;
  int*      lists  = (int*)ws;       ws += (size_t)E * T * 4;
  int*      slots  = (int*)ws;       ws += (size_t)E * T * 4;
  float*    ybuf   = (float*)ws;
  float*    out    = (float*)d_out;

  const size_t ldsRouter = (16 * AX_S + 256 * BS1_S) * sizeof(_Float16);
  const size_t ldsExpert = ((size_t)MR * AX_S + 128 * BS1_S + (size_t)H * BS2_S
                            + (size_t)MR * HH_S) * sizeof(_Float16)
                           + 2 * MR * sizeof(int);

  moe_cvt_f16     <<<(T * H) / 256, 256, 0, stream>>>(x, xh);
  moe_router_gemm1<<<dim3(T / 16, H / 256), 256, ldsRouter, stream>>>(xh, Wr1, br1, hh16);
  moe_router_topk <<<T / 256, 256, 0, stream>>>(hh16, Wr2, br2, topi, topw);
  moe_build_lists <<<E, 32, 0, stream>>>(topi, counts, lists, slots);
  moe_expert_ffn  <<<E * MT64, 1024, ldsExpert, stream>>>(xh, W1, b1, W2, b2,
                                                          counts, lists, slots, ybuf);
  moe_combine     <<<(T * H) / 256, 256, 0, stream>>>(ybuf, topw, out);
}
